// MultiMamba_43181601194397
// MI455X (gfx1250) — matrix operand order repaired and checked by reading, not compiled, not run on hardware
//
#include <hip/hip_runtime.h>

typedef __bf16 bf16;
typedef __attribute__((ext_vector_type(16))) __bf16 v16bf;
typedef __attribute__((ext_vector_type(8)))  float  v8f;

#define DEV_INLINE __device__ __forceinline__

constexpr int B_    = 16;
constexpr int L_    = 196;
constexpr int LP_   = 208;    // L padded to 13*16 for WMMA tiles
constexpr int DM_   = 384;
constexpr int DI_   = 768;
constexpr int DX_   = 1536;   // 2*D_INNER
constexpr int N_    = 16;
constexpr int R_    = 24;     // dt rank
constexpr int DBLP_ = 64;     // dt_rank + 2N = 56, padded to 64
constexpr int RED_  = 96;
constexpr int KD_   = 4;      // directions
constexpr int BL_   = B_ * L_;   // 3136
constexpr int BD_   = KD_ * B_;  // 64 (dir,batch) pairs

// direction scan-order -> original-l mapping (HT=WT=14)
DEV_INLINE int lmap(int dir, int j) {
  switch (dir) {
    case 0: return j;                                   // h
    case 1: return (L_ - 1) - j;                        // h_flip
    case 2: return (j % 14) * 14 + (j / 14);            // v
    default: { int jj = (L_ - 1) - j; return (jj % 14) * 14 + (jj / 14); } // v_flip
  }
}

DEV_INLINE float sigm(float x) { return 1.f / (1.f + __expf(-x)); }

// ---------------- WMMA 16x16x32 bf16 fragment helpers (wave32) ----------------
// A (16xK row-major): lane t: m=t&15, hi=t>>4; elem i: K=hi*8+i, elem 8+i: K=16+hi*8+i
DEV_INLINE v16bf load_a(const bf16* A, int lda, int m0, int k0, int lane) {
  const bf16* p = A + (size_t)(m0 + (lane & 15)) * lda + k0 + ((lane >> 4) << 3);
  v16bf a;
#pragma unroll
  for (int i = 0; i < 8; ++i) { a[i] = p[i]; a[8 + i] = p[16 + i]; }
  return a;
}
// B from N-major weights W[n][k] (== Bmat[k][n]): lane t: n=t&15, hi=t>>4; elem i: K=hi*8+i, elem 8+i: K=16+hi*8+i
DEV_INLINE v16bf load_b(const bf16* W, int ldb, int n0, int k0, int lane) {
  const bf16* p = W + (size_t)(n0 + (lane & 15)) * ldb + k0 + ((lane >> 4) << 3);
  v16bf b;
#pragma unroll
  for (int i = 0; i < 8; ++i) { b[i] = p[i]; b[8 + i] = p[16 + i]; }
  return b;
}
DEV_INLINE v8f wmma_bf16(v16bf a, v16bf b, v8f c) {
  return __builtin_amdgcn_wmma_f32_16x16x32_bf16(false, a, false, b, (short)0, c, false, false);
}
// C/D: elem r -> row m0 + r + 8*(lane>>4), col n0 + (lane&15)
DEV_INLINE void store_d_bf16(v8f c, bf16* C, int ldc, int m0, int n0, int lane) {
  int n = lane & 15, ro = (lane >> 4) * 8;
#pragma unroll
  for (int r = 0; r < 8; ++r) C[(size_t)(m0 + ro + r) * ldc + n0 + n] = (bf16)c[r];
}
DEV_INLINE void store_d_f32(v8f c, float* C, int ldc, int m0, int n0, int lane) {
  int n = lane & 15, ro = (lane >> 4) * 8;
#pragma unroll
  for (int r = 0; r < 8; ++r) C[(size_t)(m0 + ro + r) * ldc + n0 + n] = c[r];
}

// ---------------- precision conversion prepasses ----------------
__global__ __launch_bounds__(256) void k_cvt(const float* __restrict__ s, bf16* __restrict__ d, int n) {
  int i = blockIdx.x * 256 + threadIdx.x;
  if (i < n) d[i] = (bf16)s[i];
}
// xproj (K,56,768) -> (K,64,768) bf16 with zero pad rows
__global__ __launch_bounds__(256) void k_cvt_xproj(const float* __restrict__ s, bf16* __restrict__ d) {
  int i = blockIdx.x * 256 + threadIdx.x;
  if (i >= KD_ * DBLP_ * DI_) return;
  int dir = i / (DBLP_ * DI_);
  int rem = i % (DBLP_ * DI_);
  int r = rem / DI_, k = rem % DI_;
  d[i] = (r < 56) ? (bf16)s[((size_t)(dir * 56 + r)) * DI_ + k] : (bf16)0.f;
}

// ---------------- GEMM 1: xz[BL x 1536] = hidden[BL x 384] @ in_proj_w.T ----------------
__global__ __launch_bounds__(128) void k_gemm_in(const bf16* __restrict__ hs, const bf16* __restrict__ win,
                                                 bf16* __restrict__ xz) {
  int lane = threadIdx.x & 31, wv = threadIdx.x >> 5;
  int m0 = (blockIdx.x * 4 + wv) * 16;   // 196 M-tiles
  int n0 = blockIdx.y * 64;              // 24 N-groups of 64
  v8f c0 = {}, c1 = {}, c2 = {}, c3 = {};
  for (int k0 = 0; k0 < DM_; k0 += 32) {
    v16bf a = load_a(hs, DM_, m0, k0, lane);
    c0 = wmma_bf16(a, load_b(win, DM_, n0,      k0, lane), c0);
    c1 = wmma_bf16(a, load_b(win, DM_, n0 + 16, k0, lane), c1);
    c2 = wmma_bf16(a, load_b(win, DM_, n0 + 32, k0, lane), c2);
    c3 = wmma_bf16(a, load_b(win, DM_, n0 + 48, k0, lane), c3);
  }
  store_d_bf16(c0, xz, DX_, m0, n0,      lane);
  store_d_bf16(c1, xz, DX_, m0, n0 + 16, lane);
  store_d_bf16(c2, xz, DX_, m0, n0 + 32, lane);
  store_d_bf16(c3, xz, DX_, m0, n0 + 48, lane);
}

// ---------------- depthwise causal conv4 + SiLU, direction-reordered ----------------
// writes xcT[dir][b][j (LP_)][c] bf16; pad rows zeroed
__global__ __launch_bounds__(256) void k_conv(const bf16* __restrict__ xz, const float* __restrict__ cw,
                                              const float* __restrict__ cb, bf16* __restrict__ xcT) {
  int c = blockIdx.x * 256 + threadIdx.x;
  int b = blockIdx.y, dir = blockIdx.z, bd = dir * B_ + b;
  const float* w = cw + ((size_t)dir * DI_ + c) * 4;
  float w0 = w[0], w1 = w[1], w2 = w[2], w3 = w[3];
  float bias = cb[dir * DI_ + c];
  float xm3 = 0.f, xm2 = 0.f, xm1 = 0.f;
  bf16* dst = xcT + (size_t)bd * LP_ * DI_;
  for (int j = 0; j < L_; ++j) {
    int lo = lmap(dir, j);
    float xj = (float)xz[((size_t)(b * L_ + lo)) * DX_ + c];
    float v = bias + w0 * xm3 + w1 * xm2 + w2 * xm1 + w3 * xj;
    v = v * sigm(v);
    dst[(size_t)j * DI_ + c] = (bf16)v;
    xm3 = xm2; xm2 = xm1; xm1 = xj;
  }
  for (int j = L_; j < LP_; ++j) dst[(size_t)j * DI_ + c] = (bf16)0.f;
}

// ---------------- GEMM 2 (per dir,b): dbl[LP_ x 64] = xcT[LP_ x 768] @ xproj.T ----------------
__global__ __launch_bounds__(128) void k_gemm_xproj(const bf16* __restrict__ xcT, const bf16* __restrict__ xp,
                                                    float* __restrict__ dbl) {
  int lane = threadIdx.x & 31, wv = threadIdx.x >> 5;
  int mt = blockIdx.x * 4 + wv; if (mt >= 13) return;
  int bd = blockIdx.y, dir = bd >> 4;
  const bf16* A = xcT + (size_t)bd * LP_ * DI_;
  const bf16* W = xp + (size_t)dir * DBLP_ * DI_;
  float* C = dbl + (size_t)bd * LP_ * DBLP_;
  int m0 = mt * 16;
  v8f c0 = {}, c1 = {}, c2 = {}, c3 = {};
  for (int k0 = 0; k0 < DI_; k0 += 32) {
    v16bf a = load_a(A, DI_, m0, k0, lane);
    c0 = wmma_bf16(a, load_b(W, DI_, 0,  k0, lane), c0);
    c1 = wmma_bf16(a, load_b(W, DI_, 16, k0, lane), c1);
    c2 = wmma_bf16(a, load_b(W, DI_, 32, k0, lane), c2);
    c3 = wmma_bf16(a, load_b(W, DI_, 48, k0, lane), c3);
  }
  store_d_f32(c0, C, DBLP_, m0, 0,  lane);
  store_d_f32(c1, C, DBLP_, m0, 16, lane);
  store_d_f32(c2, C, DBLP_, m0, 32, lane);
  store_d_f32(c3, C, DBLP_, m0, 48, lane);
}

// ---------------- GEMM 3 (per dir,b): delta = softplus(dt @ dtproj.T + b), K=24 padded to 32 ----------------
DEV_INLINE v16bf load_b_dt(const float* dtw_dir, int col, int hi) {
  const float* wr = dtw_dir + (size_t)col * R_;
  v16bf b;
#pragma unroll
  for (int i = 0; i < 16; ++i) { int k = (i < 8) ? hi * 8 + i : 16 + hi * 8 + (i - 8); b[i] = (k < R_) ? (bf16)wr[k] : (bf16)0.f; }
  return b;
}
DEV_INLINE void store_dt(v8f c, bf16* D, const float* dtb_dir, int m0, int col0, int lane) {
  int n = lane & 15, ro = (lane >> 4) * 8;
  float bias = dtb_dir[col0 + n];
#pragma unroll
  for (int r = 0; r < 8; ++r) {
    float v = c[r] + bias;
    v = (v > 20.f) ? v : log1pf(__expf(v));      // softplus
    D[(size_t)(m0 + ro + r) * DI_ + col0 + n] = (bf16)v;
  }
}
__global__ __launch_bounds__(128) void k_gemm_dt(const float* __restrict__ dbl, const float* __restrict__ dtw,
                                                 const float* __restrict__ dtb, bf16* __restrict__ delta) {
  int lane = threadIdx.x & 31, wv = threadIdx.x >> 5;
  int mt = blockIdx.x * 4 + wv; if (mt >= 13) return;
  int nb = blockIdx.y * 64;
  int bd = blockIdx.z, dir = bd >> 4;
  int m0 = mt * 16, hi = lane >> 4;
  // A fragment: dbl columns 0..23 (dt), pad K 24..31 with zeros
  const float* arow = dbl + (size_t)bd * LP_ * DBLP_ + (size_t)(m0 + (lane & 15)) * DBLP_;
  v16bf a;
#pragma unroll
  for (int i = 0; i < 8; ++i) {
    a[i] = (bf16)arow[hi * 8 + i];                       // K = hi*8+i  (<=15, always valid)
    a[8 + i] = hi ? (bf16)0.f : (bf16)arow[16 + i];      // K = 16+hi*8+i (valid only lo half)
  }
  const float* dtwd = dtw + (size_t)dir * DI_ * R_;
  const float* dtbd = dtb + (size_t)dir * DI_;
  bf16* D = delta + (size_t)bd * LP_ * DI_;
  int nlane = lane & 15;
  v8f c0 = wmma_bf16(a, load_b_dt(dtwd, nb +      nlane, hi), v8f{});
  v8f c1 = wmma_bf16(a, load_b_dt(dtwd, nb + 16 + nlane, hi), v8f{});
  v8f c2 = wmma_bf16(a, load_b_dt(dtwd, nb + 32 + nlane, hi), v8f{});
  v8f c3 = wmma_bf16(a, load_b_dt(dtwd, nb + 48 + nlane, hi), v8f{});
  store_dt(c0, D, dtbd, m0, nb,      lane);
  store_dt(c1, D, dtbd, m0, nb + 16, lane);
  store_dt(c2, D, dtbd, m0, nb + 32, lane);
  store_dt(c3, D, dtbd, m0, nb + 48, lane);
}

// ---------------- selective scan: lanes = (channel-pair, n); 196 sequential steps ----------------
__global__ __launch_bounds__(256) void k_scan(const bf16* __restrict__ delta, const bf16* __restrict__ xcT,
                                              const float* __restrict__ dbl, const bf16* __restrict__ xz,
                                              const float* __restrict__ Alog, const float* __restrict__ Dpw,
                                              bf16* __restrict__ o) {
  int lane = threadIdx.x & 31, wv = threadIdx.x >> 5;
  int n = lane & 15;
  int c = blockIdx.x * 16 + wv * 2 + (lane >> 4);
  int b = blockIdx.y, dir = blockIdx.z, bd = dir * B_ + b;
  float A = -__expf(Alog[((size_t)dir * DI_ + c) * N_ + n]);
  float Dv = Dpw[dir * DI_ + c];
  const bf16* dlt = delta + (size_t)bd * LP_ * DI_;
  const bf16* xc  = xcT   + (size_t)bd * LP_ * DI_;
  const float* bl = dbl   + (size_t)bd * LP_ * DBLP_;
  bf16* ob = o + (size_t)bd * L_ * DI_;
  float h = 0.f;
  for (int j = 0; j < L_; ++j) {
    float dt = (float)dlt[(size_t)j * DI_ + c];
    float x  = (float)xc [(size_t)j * DI_ + c];
    float Bv = bl[(size_t)j * DBLP_ + R_ + n];
    float Cv = bl[(size_t)j * DBLP_ + R_ + N_ + n];
    float a = __expf(dt * A);
    h = a * h + (dt * x) * Bv;
    float s = h * Cv;
    s += __shfl_xor(s, 1); s += __shfl_xor(s, 2); s += __shfl_xor(s, 4); s += __shfl_xor(s, 8);
    if (n == 0) {
      int lo = lmap(dir, j);
      float z = (float)xz[((size_t)(b * L_ + lo)) * DX_ + DI_ + c];
      float y = (s + x * Dv) * (z * sigm(z));
      ob[(size_t)lo * DI_ + c] = (bf16)y;   // reverse_dir + transpose fused via scatter
    }
  }
}

// ---------------- bi_attn stage 1: per (dir,b,l) layernorm stats over 768 channels ----------------
__global__ __launch_bounds__(256) void k_stats(const bf16* __restrict__ o, float* __restrict__ mustd) {
  __shared__ float ss[256], sq[256];
  int idx = blockIdx.x;          // dirb*L + l
  int tid = threadIdx.x;
  const bf16* p = o + (size_t)idx * DI_;
  float s = 0.f, q = 0.f;
  for (int c = tid; c < DI_; c += 256) { float v = (float)p[c]; s += v; q += v * v; }
  ss[tid] = s; sq[tid] = q; __syncthreads();
  for (int st = 128; st > 0; st >>= 1) {
    if (tid < st) { ss[tid] += ss[tid + st]; sq[tid] += sq[tid + st]; }
    __syncthreads();
  }
  if (tid == 0) {
    float mu = ss[0] / DI_;
    float var = sq[0] / DI_ - mu * mu;
    mustd[idx * 2] = mu; mustd[idx * 2 + 1] = rsqrtf(var + 1e-5f);
  }
}

// ---------------- bi_attn stage 2: xg -> gelu MLP -> sigmoid gate c[dirb][c] ----------------
__global__ __launch_bounds__(768) void k_attn(const bf16* __restrict__ o, const float* __restrict__ mustd,
                                              const float* __restrict__ lng, const float* __restrict__ lnb,
                                              const float* __restrict__ grw, const float* __restrict__ grb,
                                              const float* __restrict__ csw, const float* __restrict__ csb,
                                              float* __restrict__ cbuf) {
  __shared__ float xgs[DI_];
  __shared__ float g96[RED_];
  int dirb = blockIdx.x, c = threadIdx.x;
  const bf16* ob = o + (size_t)dirb * L_ * DI_;
  const float* ms = mustd + (size_t)dirb * L_ * 2;
  float s = 0.f;
  for (int l = 0; l < L_; ++l) {
    float v = (float)ob[(size_t)l * DI_ + c];
    s += (v - ms[l * 2]) * ms[l * 2 + 1];
  }
  xgs[c] = (s / L_) * lng[c] + lnb[c];
  __syncthreads();
  if (c < RED_) {
    float t = grb[c];
    const float* wr = grw + (size_t)c * DI_;
    for (int k = 0; k < DI_; ++k) t += wr[k] * xgs[k];
    g96[c] = 0.5f * t * (1.f + erff(t * 0.70710678118f));   // exact gelu
  }
  __syncthreads();
  float u = csb[c];
  const float* wc = csw + (size_t)c * RED_;
  for (int k = 0; k < RED_; ++k) u += wc[k] * g96[k];
  cbuf[(size_t)dirb * DI_ + c] = sigm(u);
}

// ---------------- combine: acc[b,l,c] = sum_d o_d * c_d ----------------
__global__ __launch_bounds__(256) void k_combine(const bf16* __restrict__ o, const float* __restrict__ cbuf,
                                                 bf16* __restrict__ acc) {
  int i = blockIdx.x * 256 + threadIdx.x;
  if (i >= BL_ * DI_) return;
  int bl = i / DI_, c = i % DI_;
  int b = bl / L_, l = bl % L_;
  float s = 0.f;
#pragma unroll
  for (int d = 0; d < KD_; ++d) {
    int dirb = d * B_ + b;
    s += (float)o[((size_t)dirb * L_ + l) * DI_ + c] * cbuf[(size_t)dirb * DI_ + c];
  }
  acc[i] = (bf16)s;
}

// ---------------- GEMM 4: out[BL x 384] = acc[BL x 768] @ out_proj_w.T ----------------
__global__ __launch_bounds__(128) void k_gemm_out(const bf16* __restrict__ acc, const bf16* __restrict__ wout,
                                                  float* __restrict__ out) {
  int lane = threadIdx.x & 31, wv = threadIdx.x >> 5;
  int m0 = (blockIdx.x * 4 + wv) * 16;   // 196 M-tiles
  int n0 = blockIdx.y * 64;              // 6 N-groups
  v8f c0 = {}, c1 = {}, c2 = {}, c3 = {};
  for (int k0 = 0; k0 < DI_; k0 += 32) {
    v16bf a = load_a(acc, DI_, m0, k0, lane);
    c0 = wmma_bf16(a, load_b(wout, DI_, n0,      k0, lane), c0);
    c1 = wmma_bf16(a, load_b(wout, DI_, n0 + 16, k0, lane), c1);
    c2 = wmma_bf16(a, load_b(wout, DI_, n0 + 32, k0, lane), c2);
    c3 = wmma_bf16(a, load_b(wout, DI_, n0 + 48, k0, lane), c3);
  }
  store_d_f32(c0, out, DM_, m0, n0,      lane);
  store_d_f32(c1, out, DM_, m0, n0 + 16, lane);
  store_d_f32(c2, out, DM_, m0, n0 + 32, lane);
  store_d_f32(c3, out, DM_, m0, n0 + 48, lane);
}

extern "C" void kernel_launch(void* const* d_in, const int* in_sizes, int n_in,
                              void* d_out, int out_size, void* d_ws, size_t ws_size,
                              hipStream_t stream) {
  (void)in_sizes; (void)n_in; (void)out_size; (void)ws_size;
  const float* hs   = (const float*)d_in[0];
  const float* win  = (const float*)d_in[1];
  const float* wout = (const float*)d_in[2];
  const float* Alog = (const float*)d_in[3];
  const float* cw   = (const float*)d_in[4];
  const float* cb   = (const float*)d_in[5];
  const float* xpw  = (const float*)d_in[6];
  const float* dtw  = (const float*)d_in[7];
  const float* dtb  = (const float*)d_in[8];
  const float* Dpv  = (const float*)d_in[9];
  const float* lng  = (const float*)d_in[10];
  const float* lnb  = (const float*)d_in[11];
  const float* grw  = (const float*)d_in[12];
  const float* grb  = (const float*)d_in[13];
  const float* csw  = (const float*)d_in[14];
  const float* csb  = (const float*)d_in[15];

  char* ws = (char*)d_ws;
  size_t off = 0;
  auto al = [&](size_t bytes) -> char* {
    char* p = ws + off; off = (off + bytes + 255) & ~(size_t)255; return p;
  };
  bf16*  hs_bf   = (bf16*) al((size_t)BL_ * DM_ * 2);
  bf16*  win_bf  = (bf16*) al((size_t)DX_ * DM_ * 2);
  bf16*  wout_bf = (bf16*) al((size_t)DM_ * DI_ * 2);
  bf16*  xp_bf   = (bf16*) al((size_t)KD_ * DBLP_ * DI_ * 2);
  bf16*  xz_bf   = (bf16*) al((size_t)BL_ * DX_ * 2);
  bf16*  xcT     = (bf16*) al((size_t)BD_ * LP_ * DI_ * 2);
  float* dbl     = (float*)al((size_t)BD_ * LP_ * DBLP_ * 4);
  bf16*  delta   = (bf16*) al((size_t)BD_ * LP_ * DI_ * 2);
  bf16*  obuf    = (bf16*) al((size_t)BD_ * L_ * DI_ * 2);
  float* mustd   = (float*)al((size_t)BD_ * L_ * 2 * 4);
  float* cbuf    = (float*)al((size_t)BD_ * DI_ * 4);
  bf16*  accbf   = (bf16*) al((size_t)BL_ * DI_ * 2);

  k_cvt<<<(BL_ * DM_ + 255) / 256, 256, 0, stream>>>(hs, hs_bf, BL_ * DM_);
  k_cvt<<<(DX_ * DM_ + 255) / 256, 256, 0, stream>>>(win, win_bf, DX_ * DM_);
  k_cvt<<<(DM_ * DI_ + 255) / 256, 256, 0, stream>>>(wout, wout_bf, DM_ * DI_);
  k_cvt_xproj<<<(KD_ * DBLP_ * DI_ + 255) / 256, 256, 0, stream>>>(xpw, xp_bf);

  k_gemm_in<<<dim3(49, 24), 128, 0, stream>>>(hs_bf, win_bf, xz_bf);
  k_conv<<<dim3(3, 16, 4), 256, 0, stream>>>(xz_bf, cw, cb, xcT);
  k_gemm_xproj<<<dim3(4, 64), 128, 0, stream>>>(xcT, xp_bf, dbl);
  k_gemm_dt<<<dim3(4, 12, 64), 128, 0, stream>>>(dbl, dtw, dtb, delta);
  k_scan<<<dim3(48, 16, 4), 256, 0, stream>>>(delta, xcT, dbl, xz_bf, Alog, Dpv, obuf);
  k_stats<<<BD_ * L_, 256, 0, stream>>>(obuf, mustd);
  k_attn<<<BD_, 768, 0, stream>>>(obuf, mustd, lng, lnb, grw, grb, csw, csb, cbuf);
  k_combine<<<(BL_ * DI_ + 255) / 256, 256, 0, stream>>>(obuf, cbuf, accbf);
  k_gemm_out<<<dim3(49, 6), 128, 0, stream>>>(accbf, wout_bf, (float*)d_out);
}
